// SelfAttention_58274116272267
// MI455X (gfx1250) — matrix operand order repaired and checked by reading, not compiled, not run on hardware
//
#include <hip/hip_runtime.h>

// ---------------------------------------------------------------------------
// SAGAN-style self-attention, B=4 C=256 H=W=64 (N=4096), D=32.
//   k0a: tiled transpose+convert  x (B,C,N) f32 -> xbT (B,N,C) bf16
//   k0b: convert [Wq;Wk;Wv] -> wb (320x256) bf16
//   k1 : QKV projection, all-WMMA (bf16 x bf16 -> f32), Q pre-scaled by
//        log2(e)/sqrt(D) so flash softmax is pure exp2
//   k2 : fused flash attention (never materializes 4x4096x4096 energy)
// ---------------------------------------------------------------------------

typedef __attribute__((ext_vector_type(8)))  float  v8f;
typedef __attribute__((ext_vector_type(16))) __bf16 v16bf;
typedef __attribute__((ext_vector_type(8)))  __bf16 v8bf;

#define BATCH 4
#define CCH   256
#define NPIX  4096
#define DCH   32

__device__ __forceinline__ v8f wmma_bf16(v16bf a, v16bf b, v8f c) {
  return __builtin_amdgcn_wmma_f32_16x16x32_bf16(false, a, false, b,
                                                 (short)0, c, false, false);
}

// ---------------------------------------------------------------------------
// k0a: x (B,C,N) f32 -> xbT (B,N,C) bf16 via 32x32 LDS tile transpose.
// ---------------------------------------------------------------------------
__global__ __launch_bounds__(256) void xpose_kernel(
    const float* __restrict__ x, unsigned short* __restrict__ xbT_u)
{
  __shared__ float tile[32][33];
  const int blk = blockIdx.x;           // 4 * 8 * 128 blocks
  const int nt = blk & 127;             // 32-pixel tile
  const int ct = (blk >> 7) & 7;        // 32-channel tile
  const int b  = blk >> 10;
  const int tx = threadIdx.x & 31;
  const int ty = threadIdx.x >> 5;      // 0..7

#pragma unroll
  for (int r = 0; r < 32; r += 8) {
    int c = ct * 32 + r + ty;
    tile[r + ty][tx] = x[((size_t)b * CCH + c) * NPIX + nt * 32 + tx];
  }
  __syncthreads();
  __bf16* xbT = (__bf16*)xbT_u;
#pragma unroll
  for (int r = 0; r < 32; r += 8) {
    int n = nt * 32 + r + ty;
    xbT[((size_t)b * NPIX + n) * CCH + ct * 32 + tx] = (__bf16)tile[tx][r + ty];
  }
}

// ---------------------------------------------------------------------------
// k0b: concat(Wq[32x256], Wk[32x256], Wv[256x256]) f32 -> wb bf16 (row-major)
// ---------------------------------------------------------------------------
__global__ __launch_bounds__(256) void wcvt_kernel(
    const float* __restrict__ Wq, const float* __restrict__ Wk,
    const float* __restrict__ Wv, unsigned short* __restrict__ wb_u)
{
  const int i = blockIdx.x * 256 + threadIdx.x;   // 0 .. 320*256-1
  float v;
  if (i < 32 * 256)       v = Wq[i];
  else if (i < 64 * 256)  v = Wk[i - 32 * 256];
  else                    v = Wv[i - 64 * 256];
  ((__bf16*)wb_u)[i] = (__bf16)v;
}

// ---------------------------------------------------------------------------
// k1: one wave per 16(row)x16(pixel) output tile; 8 unrolled WMMA K-steps.
// All operand loads are contiguous bf16 (2xb128 for A, 2xb128 for B).
// rt 0-1 -> Q (wb rows 0..31), 2-3 -> K (rows 32..63), 4-19 -> V (64..319).
// Q,K stored (B,N,D) bf16; V stored (B,C,N) bf16.
// ---------------------------------------------------------------------------
__global__ __launch_bounds__(256) void qkv_proj_kernel(
    const unsigned short* __restrict__ xbT_u,
    const unsigned short* __restrict__ wb_u,
    const float* __restrict__ bq, const float* __restrict__ bk,
    const float* __restrict__ bv,
    unsigned short* __restrict__ qbuf_u,
    unsigned short* __restrict__ kbuf_u,
    unsigned short* __restrict__ vbuf_u)
{
  const int lane = threadIdx.x & 31;
  const int wid  = blockIdx.x * 8 + (threadIdx.x >> 5);
  const int b    = wid / (20 * 256);
  const int rem  = wid % (20 * 256);
  const int rt   = rem >> 8;
  const int nt   = rem & 255;
  const int m    = lane & 15;
  const int half = lane >> 4;
  const int n0   = nt * 16;

  const __bf16* wb  = (const __bf16*)wb_u;
  const __bf16* xbT = (const __bf16*)xbT_u;
  const __bf16* ap  = wb + (size_t)(rt * 16 + m) * CCH;            // A row
  const __bf16* xp  = xbT + ((size_t)b * NPIX + n0 + m) * CCH;     // B column

  v8f acc = {};
#pragma unroll
  for (int kc = 0; kc < CCH; kc += 32) {
    v16bf a;
    {
      v8bf lo = *(const v8bf*)(ap + kc + (half << 4));        // K=half*16..+7
      v8bf hi = *(const v8bf*)(ap + kc + (half << 4) + 8);    // K=half*16+8..+15
#pragma unroll
      for (int i = 0; i < 8; ++i) { a[i] = lo[i]; a[8 + i] = hi[i]; }
    }
    v16bf bm = *(const v16bf*)(xp + kc + (half << 4));        // K=half*16+i
    acc = wmma_bf16(a, bm, acc);
  }

  const float* bias; int r0w; int kind;
  if (rt < 2)      { bias = bq; r0w = rt * 16;       kind = 0; }
  else if (rt < 4) { bias = bk; r0w = (rt - 2) * 16; kind = 1; }
  else             { bias = bv; r0w = (rt - 4) * 16; kind = 2; }

  // log2(e)/sqrt(32): softmax becomes pure exp2 in the flash kernel
  const float qscale = 0.17677669529663688f * 1.44269504088896340736f;

  if (kind == 2) {
    __bf16* vbuf = (__bf16*)vbuf_u;                 // (B, C, N)
#pragma unroll
    for (int r = 0; r < 8; ++r) {
      int row = r0w + (half << 3) + r;
      vbuf[((size_t)b * CCH + row) * NPIX + n0 + m] = (__bf16)(acc[r] + bias[row]);
    }
  } else {
    __bf16* dst = (__bf16*)((kind == 0) ? qbuf_u : kbuf_u);   // (B, N, D)
    const float s = (kind == 0) ? qscale : 1.0f;
    v8bf pk;
#pragma unroll
    for (int r = 0; r < 8; ++r) {
      int row = r0w + (half << 3) + r;              // d index 0..31
      pk[r] = (__bf16)((acc[r] + bias[row]) * s);
    }
    size_t base = ((size_t)b * NPIX + n0 + m) * DCH + r0w + (half << 3);
    *(v8bf*)(dst + base) = pk;                      // one b128 store
  }
}

// ---------------------------------------------------------------------------
// k2: flash attention. Wave = 16 queries x 64 channels; j in chunks of 64
// (4 S-tiles -> one softmax update -> 2 P A-operands -> 8 accumulating WMMAs)
// ---------------------------------------------------------------------------
__global__ __launch_bounds__(256) void flash_attn_kernel(
    const float* __restrict__ x, const float* __restrict__ gamma,
    const unsigned short* __restrict__ qbuf_u,
    const unsigned short* __restrict__ kbuf_u,
    const unsigned short* __restrict__ vbuf_u,
    float* __restrict__ out)
{
  __shared__ __bf16 lds_p[8][16][80];   // per-wave P; 160B row stride (16B mult)

  const __bf16* qbuf = (const __bf16*)qbuf_u;
  const __bf16* kbuf = (const __bf16*)kbuf_u;
  const __bf16* vbuf = (const __bf16*)vbuf_u;

  const int lane = threadIdx.x & 31;
  const int wv   = threadIdx.x >> 5;
  const int wid  = blockIdx.x * 8 + wv;
  const int ct4  = wid & 3;             // 64-channel quarter
  const int qt   = (wid >> 2) & 255;    // 16-query tile
  const int b    = wid >> 10;

  const int i0 = qt * 16;
  const int c0 = ct4 * 64;
  const int m    = lane & 15;
  const int half = lane >> 4;

  // resident Q A-operand (already scaled by log2e/sqrt(D))
  v16bf qa;
  {
    const __bf16* qp = qbuf + ((size_t)b * NPIX + i0 + m) * DCH;
    v8bf lo = *(const v8bf*)(qp + (half << 4));
    v8bf hi = *(const v8bf*)(qp + 8 + (half << 4));
#pragma unroll
    for (int i = 0; i < 8; ++i) { qa[i] = lo[i]; qa[8 + i] = hi[i]; }
  }

  v8f o[4] = {{}, {}, {}, {}};
  float mrow[8], lrow[8];
#pragma unroll
  for (int r = 0; r < 8; ++r) { mrow[r] = -3.0e38f; lrow[r] = 0.0f; }

  const __bf16* kb = kbuf + (size_t)b * NPIX * DCH;
  const __bf16* vb = vbuf + (size_t)b * CCH * NPIX;

  for (int j0 = 0; j0 < NPIX; j0 += 64) {
    // ---- S: 4 tiles, one WMMA each (K = D = 32) ----
    v8f s[4];
#pragma unroll
    for (int tt = 0; tt < 4; ++tt) {
      v16bf kt = *(const v16bf*)(kb + (size_t)(j0 + tt * 16 + m) * DCH + (half << 4));
      v8f z = {};
      s[tt] = wmma_bf16(qa, kt, z);
    }

    // ---- online softmax (log2 domain): row max, exp2, row sum, rescale ----
    float nm[8];
#pragma unroll
    for (int r = 0; r < 8; ++r) {
      float v = fmaxf(fmaxf(s[0][r], s[1][r]), fmaxf(s[2][r], s[3][r]));
      v = fmaxf(v, __shfl_xor(v, 1));
      v = fmaxf(v, __shfl_xor(v, 2));
      v = fmaxf(v, __shfl_xor(v, 4));
      v = fmaxf(v, __shfl_xor(v, 8));   // masks <16: stays inside half-group
      nm[r] = fmaxf(mrow[r], v);
    }
#pragma unroll
    for (int r = 0; r < 8; ++r) {
      float p0 = __builtin_amdgcn_exp2f(s[0][r] - nm[r]);
      float p1 = __builtin_amdgcn_exp2f(s[1][r] - nm[r]);
      float p2 = __builtin_amdgcn_exp2f(s[2][r] - nm[r]);
      float p3 = __builtin_amdgcn_exp2f(s[3][r] - nm[r]);
      s[0][r] = p0; s[1][r] = p1; s[2][r] = p2; s[3][r] = p3;
      float rs = (p0 + p1) + (p2 + p3);
      rs += __shfl_xor(rs, 1);
      rs += __shfl_xor(rs, 2);
      rs += __shfl_xor(rs, 4);
      rs += __shfl_xor(rs, 8);
      float corr = __builtin_amdgcn_exp2f(mrow[r] - nm[r]);
      lrow[r] = lrow[r] * corr + rs;
      mrow[r] = nm[r];
#pragma unroll
      for (int t = 0; t < 4; ++t) o[t][r] *= corr;
    }

    // ---- P (C-layout) -> A-layout via wave-private LDS ----
#pragma unroll
    for (int r = 0; r < 8; ++r) {
      int row = (half << 3) + r;
      lds_p[wv][row][m]      = (__bf16)s[0][r];
      lds_p[wv][row][16 + m] = (__bf16)s[1][r];
      lds_p[wv][row][32 + m] = (__bf16)s[2][r];
      lds_p[wv][row][48 + m] = (__bf16)s[3][r];
    }
    __builtin_amdgcn_sched_barrier(0);  // DS in-order within a wave
    v16bf pa0, pa1;
    {
      const __bf16* pr = &lds_p[wv][m][0];
      v8bf l0 = *(const v8bf*)(pr + (half << 4));
      v8bf h0 = *(const v8bf*)(pr + 8 + (half << 4));
      v8bf l1 = *(const v8bf*)(pr + 32 + (half << 4));
      v8bf h1 = *(const v8bf*)(pr + 40 + (half << 4));
#pragma unroll
      for (int i = 0; i < 8; ++i) {
        pa0[i] = l0[i]; pa0[8 + i] = h0[i];
        pa1[i] = l1[i]; pa1[8 + i] = h1[i];
      }
    }

    // ---- O += P * V : 8 accumulating WMMAs ----
#pragma unroll
    for (int t = 0; t < 4; ++t) {
      const __bf16* vp = vb + (size_t)(c0 + t * 16 + m) * NPIX + j0 + (half << 4);
      v16bf vt0 = *(const v16bf*)vp;
      v16bf vt1 = *(const v16bf*)(vp + 32);
      o[t] = wmma_bf16(pa0, vt0, o[t]);
      o[t] = wmma_bf16(pa1, vt1, o[t]);
    }
  }

  // ---- epilogue: y = gamma * O/l + x ----
  const float g = gamma[0];
  float inv[8];
#pragma unroll
  for (int r = 0; r < 8; ++r) inv[r] = g / lrow[r];

#pragma unroll
  for (int t = 0; t < 4; ++t) {
#pragma unroll
    for (int r = 0; r < 8; ++r) {
      size_t idx = ((size_t)b * CCH + c0 + t * 16 + m) * NPIX
                 + i0 + (half << 3) + r;
      out[idx] = o[t][r] * inv[r] + x[idx];
    }
  }
}

// ---------------------------------------------------------------------------
extern "C" void kernel_launch(void* const* d_in, const int* in_sizes, int n_in,
                              void* d_out, int out_size, void* d_ws, size_t ws_size,
                              hipStream_t stream) {
  const float* x     = (const float*)d_in[0];
  const float* Wq    = (const float*)d_in[1];
  const float* bq    = (const float*)d_in[2];
  const float* Wk    = (const float*)d_in[3];
  const float* bk    = (const float*)d_in[4];
  const float* Wv    = (const float*)d_in[5];
  const float* bv    = (const float*)d_in[6];
  const float* gamma = (const float*)d_in[7];
  float* out = (float*)d_out;

  // workspace: Q 1MB | K 1MB | V 8MB | xbT 8MB | wb 160KB   (all bf16)
  unsigned short* qbuf = (unsigned short*)((char*)d_ws + 0);
  unsigned short* kbuf = (unsigned short*)((char*)d_ws + (1u << 20));
  unsigned short* vbuf = (unsigned short*)((char*)d_ws + (2u << 20));
  unsigned short* xbT  = (unsigned short*)((char*)d_ws + (10u << 20));
  unsigned short* wb   = (unsigned short*)((char*)d_ws + (18u << 20));

  // 4b * 8ct * 128nt 32x32 tiles
  xpose_kernel<<<4096, 256, 0, stream>>>(x, xbT);
  // 320*256 weights / 256
  wcvt_kernel<<<320, 256, 0, stream>>>(Wq, Wk, Wv, wb);
  // 4b * 20 row-tiles * 256 pixel-tiles = 20480 waves / 8 per block
  qkv_proj_kernel<<<2560, 256, 0, stream>>>(xbT, wb, bq, bk, bv,
                                            qbuf, kbuf, vbuf);
  // 4b * 256 query-tiles * 4 channel-quarters = 4096 waves / 8 per block
  flash_attn_kernel<<<512, 256, 0, stream>>>(x, gamma, qbuf, kbuf, vbuf, out);
}
